// RelNet_2199023256174
// MI455X (gfx1250) — compile-verified, matrix operand order checked by reading, not run on hardware
//
#include <hip/hip_runtime.h>
#include <hip/hip_bf16.h>

// ---------------------------------------------------------------------------
// RelNet for MI455X (gfx1250, wave32, WMMA).
//  * g-layers: fused WMMA kernel, f16 weights cached in LDS (320KB/WGP).
//  * convs 2-5: WMMA implicit GEMM (M=pixels, K=Cin*9=576, N=Cout=64) over
//    halo-padded NHWC f16 activations; weights [co][tap][ci] f16 in LDS.
//  * conv1 (Cin=3) VALU; BN = coalesced 2-stage NHWC reduction.
// Workspace use: ~228 MB.
// ---------------------------------------------------------------------------

typedef __attribute__((ext_vector_type(16))) _Float16 v16h;
typedef __attribute__((ext_vector_type(8)))  float    v8f;
typedef __attribute__((ext_vector_type(4)))  unsigned int v4u;

#define G_DIM 256

// Load a 16x32 (MxK) WMMA f16 fragment from a row-major matrix.
// CDNA5 layout: lane L holds row (L&15); lanes 0-15 hold K = {0..7,16..23},
// lanes 16-31 hold K = {8..15,24..31} (relative to k0). Caller folds the
// lane-dependent +8 ("hi") into the pointer. B fragments (KxN) load the same
// way from the N-major (transposed) matrix.
__device__ __forceinline__ v16h ld_frag(const _Float16* p) {
    union { v16h v; v4u q[2]; } u;
    u.q[0] = *(const v4u*)(p);
    u.q[1] = *(const v4u*)(p + 16);
    return u.v;
}

// ---------------------------------------------------------------------------
// Fused relation-network g-layers (unchanged from round 1; works well).
// ---------------------------------------------------------------------------
__global__ __launch_bounds__(256) void relnet_g_fused(
    const float* __restrict__ pa, const float* __restrict__ pb,
    const float* __restrict__ pq, const _Float16* __restrict__ gwT,
    const float* __restrict__ gb1, const float* __restrict__ gb2,
    const float* __restrict__ gb3, float* __restrict__ ctx_part)
{
    extern __shared__ char smem[];
    _Float16* wts   = (_Float16*)smem;                       // 65536 halves
    _Float16* hbuf0 = (_Float16*)(smem + 131072);            // 128x256 halves
    _Float16* hbuf1 = (_Float16*)(smem + 131072 + 65536);    // 128x256 halves
    __shared__ float cpart[16][G_DIM];

    const int tid = threadIdx.x;
    const int n   = blockIdx.x >> 5;
    const int i0  = (blockIdx.x & 31) << 1;

    const float* pqn = pq + (n << 8);
    for (int idx = tid; idx < 128 * G_DIM; idx += 256) {
        const int row = idx >> 8, col = idx & 255;
        const int i = i0 + (row >> 6), j = row & 63;
        float v = pa[(((n << 6) + i) << 8) + col] +
                  pb[(((n << 6) + j) << 8) + col] + pqn[col];
        hbuf0[idx] = (_Float16)fmaxf(v, 0.0f);
    }

    const int lane = tid & 31;
    const int wv   = tid >> 5;
    const int r15  = lane & 15;
    const int hi   = (lane & 16) ? 8 : 0;

    for (int l = 0; l < 3; ++l) {
        __syncthreads();
        {
            const v4u* src = (const v4u*)(gwT + l * 65536);
            v4u* dst = (v4u*)wts;
            for (int q = tid; q < 8192; q += 256) dst[q] = src[q];
        }
        if (l < 2)
            __builtin_prefetch((const char*)(gwT + (l + 1) * 65536) + tid * 512, 0, 1);
        __syncthreads();

        const _Float16* hin  = (l == 1) ? hbuf1 : hbuf0;
        _Float16*       hout = (l == 0) ? hbuf1 : hbuf0;
        const float*    gb   = (l == 0) ? gb1 : (l == 1) ? gb2 : gb3;
        const _Float16* arow = hin + ((wv << 4) + r15) * G_DIM + hi;

        for (int t = 0; t < 16; ++t) {
            v8f acc = {0.f, 0.f, 0.f, 0.f, 0.f, 0.f, 0.f, 0.f};
            const _Float16* brow = wts + ((t << 4) + r15) * G_DIM + hi;
#pragma unroll
            for (int k0 = 0; k0 < G_DIM; k0 += 32) {
                v16h A = ld_frag(arow + k0);
                v16h B = ld_frag(brow + k0);
                acc = __builtin_amdgcn_wmma_f32_16x16x32_f16(
                    false, A, false, B, (short)0, acc, false, false);
            }
            const float bias = gb[(t << 4) + r15];
            const int col = (t << 4) + r15;
            if (l < 2) {
#pragma unroll
                for (int v = 0; v < 8; ++v) {
                    float val = fmaxf(acc[v] + bias, 0.0f);
                    hout[((wv << 4) + hi + v) * G_DIM + col] = (_Float16)val;
                }
            } else {
                float p = 0.0f;
#pragma unroll
                for (int v = 0; v < 8; ++v) p += fmaxf(acc[v] + bias, 0.0f);
                cpart[(wv << 1) + (hi >> 3)][col] = p;
            }
        }
    }
    __syncthreads();
    float s = 0.0f;
#pragma unroll
    for (int r = 0; r < 16; ++r) s += cpart[r][tid];
    ctx_part[(blockIdx.x << 8) + tid] = s;
}

// ---------------------------------------------------------------------------
// WMMA implicit-GEMM conv 3x3 stride-2 over halo-padded NHWC f16 input.
// Grid: (32 images, slices). WG = 8 waves; each wave owns 16-pixel M-tiles.
// K = 9 taps x 64 ci; N = 64 co (4 N-tiles). 72 WMMAs per M-tile.
// Dynamic LDS: 36864 halves of [co][tap][ci] weights (72 KB).
// ---------------------------------------------------------------------------
__global__ __launch_bounds__(256) void conv_wmma(
    const _Float16* __restrict__ inP,   // padded NHWC f16, (Wp x Wp x 64)/img
    const _Float16* __restrict__ wT,    // [64][9][64] f16
    const float* __restrict__ bias,
    float* __restrict__ out,            // f32 NHWC (HW x 64)/img
    int Wout, int lgW, int T)           // T = Hout*Wout/16 tiles
{
    extern __shared__ char cmem[];
    _Float16* wl = (_Float16*)cmem;     // 36864 halves
    {
        const v4u* src = (const v4u*)wT;
        v4u* dst = (v4u*)wl;
        for (int q = threadIdx.x; q < 4608; q += 256) dst[q] = src[q];
    }
    __syncthreads();

    const int tid  = threadIdx.x;
    const int lane = tid & 31;
    const int wv   = tid >> 5;
    const int r15  = lane & 15;
    const int hi   = (lane & 16) ? 8 : 0;

    const int Wp = 2 * Wout + 2;
    const int HW = Wout << lgW;                // Hout*Wout (square)
    const _Float16* inN = inP + (size_t)blockIdx.x * Wp * Wp * 64;
    const _Float16* wlb = wl + r15 * 576 + hi; // per-lane B base

    for (int t = (int)blockIdx.y * 8 + wv; t < T; t += (int)gridDim.y * 8) {
        const int p  = (t << 4) + r15;         // this lane's A-row pixel
        const int py = p >> lgW;
        const int px = p & (Wout - 1);

        v8f acc[4];
#pragma unroll
        for (int nt = 0; nt < 4; ++nt)
            acc[nt] = (v8f){0.f, 0.f, 0.f, 0.f, 0.f, 0.f, 0.f, 0.f};

#pragma unroll
        for (int ky = 0; ky < 3; ++ky) {
#pragma unroll
            for (int kx = 0; kx < 3; ++kx) {
                const int tap = ky * 3 + kx;
                const _Float16* abase =
                    inN + ((2 * py + ky) * Wp + 2 * px + kx) * 64 + hi;
#pragma unroll
                for (int k0 = 0; k0 < 64; k0 += 32) {
                    v16h A = ld_frag(abase + k0);
#pragma unroll
                    for (int nt = 0; nt < 4; ++nt) {
                        v16h B = ld_frag(wlb + nt * (16 * 576) + tap * 64 + k0);
                        acc[nt] = __builtin_amdgcn_wmma_f32_16x16x32_f16(
                            false, A, false, B, (short)0, acc[nt], false, false);
                    }
                }
            }
        }

        float* outN = out + ((size_t)blockIdx.x * HW + (t << 4)) * 64;
#pragma unroll
        for (int nt = 0; nt < 4; ++nt) {
            const float bv = bias[nt * 16 + r15];
#pragma unroll
            for (int v = 0; v < 8; ++v)
                outN[(hi + v) * 64 + nt * 16 + r15] = acc[nt][v] + bv;
        }
    }
}

// Conv1: Cin=3, VALU, img NCHW f32 -> conv-out NHWC f32 (128x128x64 / img).
__global__ __launch_bounds__(256) void conv1_nhwc(
    const float* __restrict__ img, const float* __restrict__ w,
    const float* __restrict__ bias, float* __restrict__ out)
{
    int idx = blockIdx.x * 256 + threadIdx.x;
    if (idx >= 32 * 128 * 128 * 64) return;
    const int co = idx & 63;
    const int p  = (idx >> 6) & 16383;
    const int n  = idx >> 20;
    const int y  = p >> 7, x = p & 127;

    float s = bias[co];
#pragma unroll
    for (int ci = 0; ci < 3; ++ci) {
        const float* ip = img + ((n * 3 + ci) * 256) * 256;
        const float* wc = w + (co * 3 + ci) * 9;
#pragma unroll
        for (int ky = 0; ky < 3; ++ky) {
            const int iy = 2 * y + ky - 1;
            if ((unsigned)iy >= 256u) continue;
#pragma unroll
            for (int kx = 0; kx < 3; ++kx) {
                const int ix = 2 * x + kx - 1;
                if ((unsigned)ix >= 256u) continue;
                s += ip[iy * 256 + ix] * wc[ky * 3 + kx];
            }
        }
    }
    out[idx] = s;
}

// BN stage 1: coalesced NHWC partial sums. part[wg][0..63]=sum, [64..127]=sumsq.
__global__ __launch_bounds__(256) void bn_part_nhwc(
    const float* __restrict__ x, float* __restrict__ part, int M /*pixels*/)
{
    __shared__ float rs[256], rq[256];
    const int tid = threadIdx.x;
    const int c = tid & 63, sub = tid >> 6;
    float s = 0.f, q = 0.f;
    for (int pix = blockIdx.x * 4 + sub; pix < M; pix += gridDim.x * 4) {
        const float v = x[(size_t)pix * 64 + c];
        s += v; q += v * v;
    }
    rs[tid] = s; rq[tid] = q; __syncthreads();
    if (tid < 64) {
        float a = 0.f, b = 0.f;
        for (int g = 0; g < 4; ++g) { a += rs[g * 64 + tid]; b += rq[g * 64 + tid]; }
        part[blockIdx.x * 128 + tid] = a;
        part[blockIdx.x * 128 + 64 + tid] = b;
    }
}

// BN stage 2: finalize per-channel (scale, shift).
__global__ void bn_finalize(
    const float* __restrict__ part, int nparts,
    const float* __restrict__ gamma, const float* __restrict__ beta,
    float* __restrict__ ss, float invM)
{
    const int c = threadIdx.x;
    if (c >= 64) return;
    float s = 0.f, q = 0.f;
    for (int p = 0; p < nparts; ++p) { s += part[p * 128 + c]; q += part[p * 128 + 64 + c]; }
    const float mu = s * invM;
    const float var = q * invM - mu * mu;
    const float sc = gamma[c] * rsqrtf(var + 1e-5f);
    ss[2 * c] = sc;
    ss[2 * c + 1] = beta[c] - mu * sc;
}

// BN apply + ReLU, f32 NHWC -> halo-padded NHWC f16 (halo pre-zeroed).
__global__ __launch_bounds__(256) void bn_apply_nhwc(
    const float* __restrict__ x, const float* __restrict__ ss,
    _Float16* __restrict__ outp, int HW, int W, int lgW)
{
    int idx = blockIdx.x * 256 + threadIdx.x;
    if (idx >= 32 * HW * 64) return;
    const int c = idx & 63;
    const int pp = idx >> 6;
    const int pix = pp % HW, n = pp / HW;
    const int y = pix >> lgW, xx = pix & (W - 1);
    const int Wp = W + 2;
    const float v = fmaxf(fmaf(x[idx], ss[2 * c], ss[2 * c + 1]), 0.0f);
    outp[((n * Wp + y + 1) * Wp + xx + 1) * 64 + c] = (_Float16)v;
}

__global__ __launch_bounds__(256) void zero_u32(unsigned* __restrict__ p, int n) {
    int idx = blockIdx.x * 256 + threadIdx.x;
    if (idx < n) p[idx] = 0u;
}

// Convert g-layer weights (f32 KxN) -> f16 N-major (transposed).
__global__ __launch_bounds__(256) void cvt_gw(
    const float* __restrict__ g1, const float* __restrict__ g2,
    const float* __restrict__ g3, _Float16* __restrict__ gwT)
{
    int idx = blockIdx.x * 256 + threadIdx.x;
    if (idx >= 3 * 65536) return;
    const int l = idx >> 16, e = idx & 65535;
    const int nn = e >> 8, kk = e & 255;
    const float* g = (l == 0) ? g1 : (l == 1) ? g2 : g3;
    gwT[idx] = (_Float16)g[kk * 256 + nn];
}

// Convert conv weights OIHW f32 -> [co][tap][ci] f16 for layers 2-5.
__global__ __launch_bounds__(256) void cvt_cw(
    const float* __restrict__ w1, const float* __restrict__ w2,
    const float* __restrict__ w3, const float* __restrict__ w4,
    _Float16* __restrict__ wTc)
{
    int idx = blockIdx.x * 256 + threadIdx.x;
    if (idx >= 4 * 36864) return;
    const int l = idx / 36864, e = idx % 36864;
    const int co = e / 576, rem = e % 576;
    const int tap = rem >> 6, ci = rem & 63;
    const float* w = (l == 0) ? w1 : (l == 1) ? w2 : (l == 2) ? w3 : w4;
    wTc[idx] = (_Float16)w[(co * 64 + ci) * 9 + tap];
}

// pa / pb from padded NHWC f16 act5 (10x10x64 per image; interior 8x8).
__global__ __launch_bounds__(256) void pab_gemm(
    const _Float16* __restrict__ act5p, const float* __restrict__ gw0,
    float* __restrict__ outp, int rowoff)
{
    int idx = blockIdx.x * 256 + threadIdx.x;
    if (idx >= 32 * 64 * 256) return;
    const int d = idx & 255, o = (idx >> 8) & 63, n = idx >> 14;
    const _Float16* cell = act5p + ((n * 10 + (o >> 3) + 1) * 10 + (o & 7) + 1) * 64;
    float s = 0.0f;
    for (int c = 0; c < 64; ++c)
        s += (float)cell[c] * gw0[(rowoff + c) * 256 + d];
    s += (float)(o >> 3) * gw0[(rowoff + 64) * 256 + d];
    s += (float)(o & 7)  * gw0[(rowoff + 65) * 256 + d];
    outp[idx] = s;
}

// pq = ques^T @ Wq + b1
__global__ __launch_bounds__(256) void pq_gemm(
    const float* __restrict__ ques, const float* __restrict__ gw0,
    const float* __restrict__ gb0, float* __restrict__ outp)
{
    int idx = blockIdx.x * 256 + threadIdx.x;
    if (idx >= 32 * 256) return;
    const int n = idx >> 8, d = idx & 255;
    float s = gb0[d];
    for (int q = 0; q < 128; ++q)
        s += ques[q * 32 + n] * gw0[(132 + q) * 256 + d];
    outp[idx] = s;
}

// Head: mean-pool context, 2x (256x256 relu), 256x1000, log_softmax.
__global__ __launch_bounds__(256) void head_kernel(
    const float* __restrict__ ctx_part,
    const float* __restrict__ fw0, const float* __restrict__ fb0,
    const float* __restrict__ fw1, const float* __restrict__ fb1,
    const float* __restrict__ fw2, const float* __restrict__ fb2,
    float* __restrict__ out)
{
    __shared__ float ctx[256], y1[256], y2[256], sc[1000], red[256];
    const int n = blockIdx.x, tid = threadIdx.x;

    float s = 0.0f;
    for (int ib = 0; ib < 32; ++ib)
        s += ctx_part[((n << 5) + ib) * 256 + tid];
    ctx[tid] = s * (1.0f / 4096.0f);
    __syncthreads();

    s = fb0[tid];
    for (int k = 0; k < 256; ++k) s += ctx[k] * fw0[k * 256 + tid];
    y1[tid] = fmaxf(s, 0.0f);
    __syncthreads();

    s = fb1[tid];
    for (int k = 0; k < 256; ++k) s += y1[k] * fw1[k * 256 + tid];
    y2[tid] = fmaxf(s, 0.0f);
    __syncthreads();

    for (int v = tid; v < 1000; v += 256) {
        float a = fb2[v];
        for (int k = 0; k < 256; ++k) a += y2[k] * fw2[k * 1000 + v];
        sc[v] = a;
    }
    __syncthreads();

    float m = -3.4e38f;
    for (int v = tid; v < 1000; v += 256) m = fmaxf(m, sc[v]);
    red[tid] = m; __syncthreads();
    for (int st = 128; st > 0; st >>= 1) {
        if (tid < st) red[tid] = fmaxf(red[tid], red[tid + st]);
        __syncthreads();
    }
    const float mx = red[0];
    __syncthreads();

    float e = 0.0f;
    for (int v = tid; v < 1000; v += 256) e += expf(sc[v] - mx);
    red[tid] = e; __syncthreads();
    for (int st = 128; st > 0; st >>= 1) {
        if (tid < st) red[tid] += red[tid + st];
        __syncthreads();
    }
    const float lse = logf(red[0]);
    __syncthreads();

    for (int v = tid; v < 1000; v += 256)
        out[n * 1000 + v] = sc[v] - mx - lse;
}

// ---------------------------------------------------------------------------
extern "C" void kernel_launch(void* const* d_in, const int* in_sizes, int n_in,
                              void* d_out, int out_size, void* d_ws, size_t ws_size,
                              hipStream_t stream) {
    const float* img  = (const float*)d_in[0];
    const float* ques = (const float*)d_in[1];
    const float *cw[5], *cb[5], *bng[5], *bnb[5];
    for (int i = 0; i < 5; ++i) {
        cw[i]  = (const float*)d_in[2 + 4 * i];
        cb[i]  = (const float*)d_in[3 + 4 * i];
        bng[i] = (const float*)d_in[4 + 4 * i];
        bnb[i] = (const float*)d_in[5 + 4 * i];
    }
    const float* gw0 = (const float*)d_in[22];
    const float* gb0 = (const float*)d_in[23];
    const float* gw1 = (const float*)d_in[24];
    const float* gb1 = (const float*)d_in[25];
    const float* gw2 = (const float*)d_in[26];
    const float* gb2 = (const float*)d_in[27];
    const float* gw3 = (const float*)d_in[28];
    const float* gb3 = (const float*)d_in[29];
    const float* fw0 = (const float*)d_in[30];
    const float* fb0 = (const float*)d_in[31];
    const float* fw1 = (const float*)d_in[32];
    const float* fb1 = (const float*)d_in[33];
    const float* fw2 = (const float*)d_in[34];
    const float* fb2 = (const float*)d_in[35];
    float* out = (float*)d_out;

    // Workspace layout (~228 MB).
    float* F   = (float*)d_ws;
    float* C32 = F;                             // 33,554,432 f (conv out, NHWC)
    float* ss  = C32 + 33554432;                //        256 f
    float* bnp = ss + 256;                      //     32,768 f (BN partials)
    float* pa  = bnp + 32768;                   //    524,288 f
    float* pb  = pa + 524288;                   //    524,288 f
    float* pqv = pb + 524288;                   //      8,192 f
    float* cxp = pqv + 8192;                    //    262,144 f
    _Float16* gwT = (_Float16*)(cxp + 262144);  //    196,608 h (g weights)
    _Float16* wTc = gwT + 196608;               //    147,456 h (conv weights)
    _Float16* PH0 = wTc + 147456;               // 34,611,200 h (padded acts)
    _Float16* PH1 = PH0 + 34611200;             //  8,921,088 h (padded acts)

    const dim3 B(256);

    cvt_gw<<<768, B, 0, stream>>>(gw1, gw2, gw3, gwT);
    cvt_cw<<<576, B, 0, stream>>>(cw[1], cw[2], cw[3], cw[4], wTc);

    // BN pipeline helper: stats on f32 NHWC in C32, apply into padded f16 dst.
    auto bn_nhwc = [&](int W, int lgW, const float* gamma, const float* beta,
                       _Float16* dstPad) {
        const int HW = W * W, Mpix = 32 * HW;
        bn_part_nhwc<<<256, B, 0, stream>>>(C32, bnp, Mpix);
        bn_finalize<<<1, 64, 0, stream>>>(bnp, 256, gamma, beta, ss,
                                          1.0f / (float)Mpix);
        const int padded_halves = 32 * (W + 2) * (W + 2) * 64;
        zero_u32<<<(padded_halves / 2 + 255) / 256, B, 0, stream>>>(
            (unsigned*)dstPad, padded_halves / 2);
        const int total = Mpix * 64;
        bn_apply_nhwc<<<(total + 255) / 256, B, 0, stream>>>(C32, ss, dstPad,
                                                             HW, W, lgW);
    };

    // ---- conv tower ----
    conv1_nhwc<<<131072, B, 0, stream>>>(img, cw[0], cb[0], C32);
    bn_nhwc(128, 7, bng[0], bnb[0], PH0);

    const size_t cw_smem = 36864 * 2;  // 72 KB dynamic LDS
    // conv2: 64x64 out, T=256 tiles
    conv_wmma<<<dim3(32, 32), B, cw_smem, stream>>>(PH0, wTc + 0 * 36864, cb[1],
                                                    C32, 64, 6, 256);
    bn_nhwc(64, 6, bng[1], bnb[1], PH1);
    // conv3: 32x32 out, T=64
    conv_wmma<<<dim3(32, 8), B, cw_smem, stream>>>(PH1, wTc + 1 * 36864, cb[2],
                                                   C32, 32, 5, 64);
    bn_nhwc(32, 5, bng[2], bnb[2], PH0);
    // conv4: 16x16 out, T=16
    conv_wmma<<<dim3(32, 2), B, cw_smem, stream>>>(PH0, wTc + 2 * 36864, cb[3],
                                                   C32, 16, 4, 16);
    bn_nhwc(16, 4, bng[3], bnb[3], PH1);
    // conv5: 8x8 out, T=4
    conv_wmma<<<dim3(32, 1), B, cw_smem, stream>>>(PH1, wTc + 3 * 36864, cb[4],
                                                   C32, 8, 3, 4);
    bn_nhwc(8, 3, bng[4], bnb[4], PH0);   // act5 (padded 10x10x64) in PH0

    // ---- relation network ----
    pab_gemm<<<2048, B, 0, stream>>>(PH0, gw0, pa, 0);
    pab_gemm<<<2048, B, 0, stream>>>(PH0, gw0, pb, 66);
    pq_gemm<<<32, B, 0, stream>>>(ques, gw0, gb0, pqv);

    const size_t g_smem = 131072 + 65536 + 65536;  // 256 KB dynamic LDS
    relnet_g_fused<<<1024, B, g_smem, stream>>>(pa, pb, pqv, gwT,
                                                gb1, gb2, gb3, cxp);

    head_kernel<<<32, B, 0, stream>>>(cxp, fw0, fb0, fw1, fb1, fw2, fb2, out);
}
